// Net_33432025432124
// MI455X (gfx1250) — matrix operand order repaired and checked by reading, not compiled, not run on hardware
//
#include <hip/hip_runtime.h>

#define KBINS 125
#define NNODES 10000
#define NEDGES 160000

typedef unsigned short u16;
typedef unsigned int   u32;
typedef __attribute__((ext_vector_type(16))) __bf16 v16bf;
typedef __attribute__((ext_vector_type(8)))  float  v8f;

__device__ inline u16 f2bf(float f) {
  union { float f; u32 u; } v; v.f = f;
  u32 r = v.u + 0x7FFFu + ((v.u >> 16) & 1u);
  return (u16)(r >> 16);
}
__device__ inline float bf2f(u16 b) {
  union { u32 u; float f; } v; v.u = ((u32)b) << 16;
  return v.f;
}

// ---------------- spline basis ----------------
__global__ void basis_kernel(const float* __restrict__ pseudo,
                             float* __restrict__ basis, int* __restrict__ bidx) {
  int e = blockIdx.x * blockDim.x + threadIdx.x;
  if (e >= NEDGES) return;
  float fr[3]; int lo[3];
#pragma unroll
  for (int d = 0; d < 3; ++d) {
    float p = pseudo[e * 3 + d] * 4.0f;   // K-1 = 4
    float l = floorf(p);
    l = fminf(fmaxf(l, 0.f), 3.f);        // clip to K-2
    fr[d] = p - l;
    lo[d] = (int)l;
  }
  const int strides[3] = {25, 5, 1};
#pragma unroll
  for (int b = 0; b < 8; ++b) {
    float w = 1.f; int idx = 0;
#pragma unroll
    for (int d = 0; d < 3; ++d) {
      int bit = (b >> d) & 1;
      w *= bit ? fr[d] : (1.f - fr[d]);
      idx += (lo[d] + bit) * strides[d];
    }
    basis[e * 8 + b] = w;
    bidx[e * 8 + b]  = idx;
  }
}

// ---------------- degree ----------------
__global__ void deg_kernel(const int* __restrict__ dst, float* __restrict__ deg) {
  int e = blockIdx.x * blockDim.x + threadIdx.x;
  if (e >= NEDGES) return;
  unsafeAtomicAdd(&deg[dst[e]], 1.0f);
}

// ---------------- f32 -> bf16 convert ----------------
__global__ void cvt_kernel(const float* __restrict__ in, u16* __restrict__ out, int n) {
  int t = blockIdx.x * blockDim.x + threadIdx.x;
  if (t < n) out[t] = f2bf(in[t]);
}

// ------------- f32 [Kd x Nd] row-major -> bf16 transposed [Nd x Kd] -------------
__global__ void cvtT_kernel(const float* __restrict__ in, u16* __restrict__ out,
                            int Kd, int Nd) {
  int t = blockIdx.x * blockDim.x + threadIdx.x;
  if (t >= Kd * Nd) return;
  int k = t / Nd, n = t % Nd;
  out[(size_t)n * Kd + k] = f2bf(in[t]);
}

// ---------------- layer-1 Z (ci==1, outer product) ----------------
__global__ void z1_kernel(const float* __restrict__ x, const float* __restrict__ w1,
                          u16* __restrict__ Z, int total) {  // total = N*125*32
  int t = blockIdx.x * blockDim.x + threadIdx.x;
  if (t >= total) return;
  int n = t / (KBINS * 32);
  int j = t % (KBINS * 32);
  Z[t] = f2bf(x[n] * w1[j]);
}

// -------- weight transpose: W[k,c,o] -> Wt[(k*co+o), c]  (B^T, K-major, bf16) --------
__global__ void wt_kernel(const float* __restrict__ W, u16* __restrict__ Wt,
                          int ci, int co, int total) {
  int t = blockIdx.x * blockDim.x + threadIdx.x;
  if (t >= total) return;
  int k   = t / (ci * co);
  int rem = t % (ci * co);
  int c   = rem / co;
  int o   = rem % co;
  Wt[(size_t)(k * co + o) * ci + c] = f2bf(W[t]);
}

// ---------------- WMMA GEMM: C[M x Nd] = A[M x Kd] (bf16) * B[Kd x Nd] ----------------
// B supplied TRANSPOSED (Bt[Nd x Kd], K-major) so fragments load as b128.
// One wave computes a 16x64 strip (4 N-tiles), reusing its A fragment 4x.
template <int STORE_BF16, int DO_ELU, int HAS_BIAS>
__global__ __launch_bounds__(128)
void wmma_gemm(const u16* __restrict__ A, const u16* __restrict__ Bt,
               const float* __restrict__ bias, void* __restrict__ Cv,
               int M, int Kd, int Nd) {
  int strips_n = Nd >> 6;                      // 64-wide strips
  int total    = (M >> 4) * strips_n;
  int wave = blockIdx.x * (blockDim.x >> 5) + (threadIdx.x >> 5);
  if (wave >= total) return;
  int mt = wave / strips_n, nq = wave % strips_n;
  int lane  = threadIdx.x & 31;
  int mrow  = (mt << 4) + (lane & 15);
  int ncol0 = (nq << 6) + (lane & 15);
  int koffA = (lane >> 4) << 4;   // A: hi lanes hold K 16..31
  int koffB = (lane >> 4) << 4;   // B: hi lanes hold K 16..31

  v8f acc0 = {}, acc1 = {}, acc2 = {}, acc3 = {};
  union Frag { v16bf v; uint4 q[2]; } a, b0, b1, b2, b3;
  const u16* arow = A + (size_t)mrow * Kd;
  const u16* brow0 = Bt + (size_t)(ncol0)      * Kd;
  const u16* brow1 = Bt + (size_t)(ncol0 + 16) * Kd;
  const u16* brow2 = Bt + (size_t)(ncol0 + 32) * Kd;
  const u16* brow3 = Bt + (size_t)(ncol0 + 48) * Kd;

  for (int k0 = 0; k0 < Kd; k0 += 32) {
    a.q[0] = *(const uint4*)(arow + k0 + koffA);        // K = koffA + 0..7
    a.q[1] = *(const uint4*)(arow + k0 + koffA + 8);    // K = koffA + 8..15
    b0.q[0] = *(const uint4*)(brow0 + k0 + koffB);
    b0.q[1] = *(const uint4*)(brow0 + k0 + koffB + 8);
    b1.q[0] = *(const uint4*)(brow1 + k0 + koffB);
    b1.q[1] = *(const uint4*)(brow1 + k0 + koffB + 8);
    b2.q[0] = *(const uint4*)(brow2 + k0 + koffB);
    b2.q[1] = *(const uint4*)(brow2 + k0 + koffB + 8);
    b3.q[0] = *(const uint4*)(brow3 + k0 + koffB);
    b3.q[1] = *(const uint4*)(brow3 + k0 + koffB + 8);
    acc0 = __builtin_amdgcn_wmma_f32_16x16x32_bf16(false, a.v, false, b0.v, (short)0, acc0, false, false);
    acc1 = __builtin_amdgcn_wmma_f32_16x16x32_bf16(false, a.v, false, b1.v, (short)0, acc1, false, false);
    acc2 = __builtin_amdgcn_wmma_f32_16x16x32_bf16(false, a.v, false, b2.v, (short)0, acc2, false, false);
    acc3 = __builtin_amdgcn_wmma_f32_16x16x32_bf16(false, a.v, false, b3.v, (short)0, acc3, false, false);
  }

  int rbase = (mt << 4) + ((lane >> 4) << 3);  // C: VGPR r -> M = r (+8 for hi lanes)
  v8f* accs[4] = {&acc0, &acc1, &acc2, &acc3};
#pragma unroll
  for (int j = 0; j < 4; ++j) {
    int ncol = ncol0 + 16 * j;
    float bb = HAS_BIAS ? bias[ncol] : 0.f;
#pragma unroll
    for (int r = 0; r < 8; ++r) {
      float vv = (*accs[j])[r] + bb;
      if (DO_ELU) vv = vv > 0.f ? vv : (__expf(vv) - 1.f);
      size_t ci_ = (size_t)(rbase + r) * Nd + ncol;
      if (STORE_BF16) ((u16*)Cv)[ci_] = f2bf(vv);
      else            ((float*)Cv)[ci_] = vv;
    }
  }
}

// ---------------- edge gather + mean-scatter ----------------
__global__ __launch_bounds__(256)
void edge_gather(const u16* __restrict__ Z, const float* __restrict__ basis,
                 const int* __restrict__ bidx, const int* __restrict__ src,
                 const int* __restrict__ dst, float* __restrict__ agg, int co) {
  int e = blockIdx.x * 8 + (threadIdx.x >> 5);
  if (e >= NEDGES) return;
  int lane = threadIdx.x & 31;
  int s = src[e], d = dst[e];
  float a0 = 0.f, a1 = 0.f;
  if (co == 64) {
#pragma unroll
    for (int b = 0; b < 8; ++b) {
      float w = basis[e * 8 + b];
      int  kk = bidx[e * 8 + b];
      const u16* z = Z + ((size_t)s * KBINS + kk) * 64;
      u32 p = *(const u32*)(z + 2 * lane);            // two consecutive bf16
      a0 += w * bf2f((u16)(p & 0xFFFFu));
      a1 += w * bf2f((u16)(p >> 16));
    }
    unsafeAtomicAdd(&agg[(size_t)d * 64 + 2 * lane],     a0);
    unsafeAtomicAdd(&agg[(size_t)d * 64 + 2 * lane + 1], a1);
  } else {  // co == 32
#pragma unroll
    for (int b = 0; b < 8; ++b) {
      float w = basis[e * 8 + b];
      int  kk = bidx[e * 8 + b];
      const u16* z = Z + ((size_t)s * KBINS + kk) * 32;
      a0 += w * bf2f(z[lane]);
    }
    unsafeAtomicAdd(&agg[(size_t)d * 32 + lane], a0);
  }
}

// ---------------- node update: mean + root matvec + bias + ELU -> bf16 ----------------
__global__ void node_update(const float* __restrict__ agg, const float* __restrict__ deg,
                            const u16* __restrict__ xin, const float* __restrict__ root,
                            const float* __restrict__ bias, u16* __restrict__ xout,
                            int ci, int co, int total) {
  int t = blockIdx.x * blockDim.x + threadIdx.x;
  if (t >= total) return;
  int n = t / co, o = t % co;
  const u16* xr = xin + (size_t)n * ci;
  float r = 0.f;
  for (int c = 0; c < ci; ++c) r += bf2f(xr[c]) * root[c * co + o];
  float v = agg[t] / fmaxf(deg[n], 1.0f) + r + bias[o];
  v = v > 0.f ? v : (__expf(v) - 1.f);
  xout[t] = f2bf(v);
}

// ---------------- in-place log-softmax over 1024 cols ----------------
__global__ __launch_bounds__(256)
void log_softmax_kernel(float* __restrict__ out) {
  __shared__ float red[256];
  float* row = out + (size_t)blockIdx.x * 1024;
  int t = threadIdx.x;
  float m = -1e30f;
  for (int c = t; c < 1024; c += 256) m = fmaxf(m, row[c]);
  red[t] = m; __syncthreads();
  for (int s = 128; s > 0; s >>= 1) { if (t < s) red[t] = fmaxf(red[t], red[t + s]); __syncthreads(); }
  m = red[0]; __syncthreads();
  float ss = 0.f;
  for (int c = t; c < 1024; c += 256) ss += __expf(row[c] - m);
  red[t] = ss; __syncthreads();
  for (int s = 128; s > 0; s >>= 1) { if (t < s) red[t] += red[t + s]; __syncthreads(); }
  float lse = m + __logf(red[0]);
  __syncthreads();
  for (int c = t; c < 1024; c += 256) row[c] -= lse;
}

static inline int cdiv(long long a, long long b) { return (int)((a + b - 1) / b); }

extern "C" void kernel_launch(void* const* d_in, const int* in_sizes, int n_in,
                              void* d_out, int out_size, void* d_ws, size_t ws_size,
                              hipStream_t stream) {
  (void)in_sizes; (void)n_in; (void)out_size; (void)ws_size;
  const float* x      = (const float*)d_in[0];
  const int*   eidx   = (const int*)d_in[1];
  const float* pseudo = (const float*)d_in[2];
  const float *W[6], *root[6], *bias[6];
  for (int i = 0; i < 6; ++i) {
    W[i]    = (const float*)d_in[3 + 3 * i];
    root[i] = (const float*)d_in[4 + 3 * i];
    bias[i] = (const float*)d_in[5 + 3 * i];
  }
  const float* fc1w = (const float*)d_in[21];
  const float* fc1b = (const float*)d_in[22];
  const float* fc2w = (const float*)d_in[23];
  const float* fc2b = (const float*)d_in[24];
  const int* src = eidx;
  const int* dst = eidx + NEDGES;

  // ---- workspace carve ----
  char* ws = (char*)d_ws;
  size_t off = 0;
  auto carve = [&](size_t bytes) -> void* {
    void* p = ws + off;
    off = (off + bytes + 255) & ~(size_t)255;
    return p;
  };
  float* basis = (float*)carve((size_t)NEDGES * 8 * 4);
  int*   bidx  = (int*)  carve((size_t)NEDGES * 8 * 4);
  float* deg   = (float*)carve((size_t)NNODES * 4);
  u16*   hx    = (u16*)  carve((size_t)NNODES * 2);            // layer-1 input bf16
  u16*   hA    = (u16*)  carve((size_t)NNODES * 64 * 2);
  u16*   hB    = (u16*)  carve((size_t)NNODES * 64 * 2);
  float* agg   = (float*)carve((size_t)NNODES * 64 * 4);
  u16*   Wt    = (u16*)  carve((size_t)KBINS * 64 * 64 * 2);
  u16*   fc1wb = (u16*)  carve((size_t)64 * 256 * 2);
  u16*   fc2wb = (u16*)  carve((size_t)256 * 1024 * 2);
  u16*   hfc1  = (u16*)  carve((size_t)NNODES * 256 * 2);
  u16*   Z     = (u16*)  carve((size_t)NNODES * KBINS * 64 * 2);

  // ---- edge preprocessing (basis, indices, degrees) ----
  basis_kernel<<<cdiv(NEDGES, 256), 256, 0, stream>>>(pseudo, basis, bidx);
  hipMemsetAsync(deg, 0, (size_t)NNODES * 4, stream);
  deg_kernel<<<cdiv(NEDGES, 256), 256, 0, stream>>>(dst, deg);
  cvt_kernel<<<cdiv(NNODES, 256), 256, 0, stream>>>(x, hx, NNODES);

  // ---- layer 1 (ci=1, co=32): Z via outer product ----
  {
    int totalZ = NNODES * KBINS * 32;
    z1_kernel<<<cdiv(totalZ, 256), 256, 0, stream>>>(x, W[0], Z, totalZ);
    hipMemsetAsync(agg, 0, (size_t)NNODES * 32 * 4, stream);
    edge_gather<<<cdiv(NEDGES, 8), 256, 0, stream>>>(Z, basis, bidx, src, dst, agg, 32);
    node_update<<<cdiv(NNODES * 32, 256), 256, 0, stream>>>(
        agg, deg, hx, root[0], bias[0], hA, 1, 32, NNODES * 32);
  }

  // ---- layers 2..6: WMMA Z-GEMM + gather + update ----
  u16* hcur = hA; u16* hnext = hB;
  const int cis[6]  = {1, 32, 64, 64, 64, 64};
  const int cos_[6] = {32, 64, 64, 64, 64, 64};
  for (int l = 1; l < 6; ++l) {
    int ci = cis[l], co = cos_[l];
    int wtot = KBINS * ci * co;
    wt_kernel<<<cdiv(wtot, 256), 256, 0, stream>>>(W[l], Wt, ci, co, wtot);
    int Nd = KBINS * co;                       // 8000 (multiple of 64)
    int strips = (NNODES / 16) * (Nd / 64);
    wmma_gemm<1, 0, 0><<<cdiv(strips, 4), 128, 0, stream>>>(
        hcur, Wt, nullptr, Z, NNODES, ci, Nd);
    hipMemsetAsync(agg, 0, (size_t)NNODES * co * 4, stream);
    edge_gather<<<cdiv(NEDGES, 8), 256, 0, stream>>>(Z, basis, bidx, src, dst, agg, co);
    node_update<<<cdiv(NNODES * co, 256), 256, 0, stream>>>(
        agg, deg, hcur, root[l], bias[l], hnext, ci, co, NNODES * co);
    u16* tmp = hcur; hcur = hnext; hnext = tmp;
  }

  // ---- fc1: [N,64] @ [64,256] + bias, ELU -> bf16 ----
  cvtT_kernel<<<cdiv(64 * 256, 256), 256, 0, stream>>>(fc1w, fc1wb, 64, 256);
  {
    int strips = (NNODES / 16) * (256 / 64);
    wmma_gemm<1, 1, 1><<<cdiv(strips, 4), 128, 0, stream>>>(
        hcur, fc1wb, fc1b, hfc1, NNODES, 64, 256);
  }
  // ---- fc2: [N,256] @ [256,1024] + bias -> f32 logits in d_out ----
  cvtT_kernel<<<cdiv(256 * 1024, 256), 256, 0, stream>>>(fc2w, fc2wb, 256, 1024);
  {
    int strips = (NNODES / 16) * (1024 / 64);
    wmma_gemm<0, 0, 1><<<cdiv(strips, 4), 128, 0, stream>>>(
        hfc1, fc2wb, fc2b, (float*)d_out, NNODES, 256, 1024);
  }
  // ---- log-softmax in place ----
  log_softmax_kernel<<<NNODES, 256, 0, stream>>>((float*)d_out);
}
